// SkeletonEmbeddingLoss_10565619548450
// MI455X (gfx1250) — compile-verified, matrix operand order checked by reading, not run on hardware
//
#include <hip/hip_runtime.h>

#define B_ 16
#define S_ 2
#define H_ 768
#define W_ 768
#define K_ 16
#define HW_ (H_ * W_)

typedef __attribute__((ext_vector_type(2)))  float    v2f;
typedef __attribute__((ext_vector_type(8)))  float    v8f;
typedef __attribute__((ext_vector_type(16))) _Float16 v16h;

// ---------------------------------------------------------------------------
// Bilinear gather with zeros padding (matches reference: clamp index, mask OOB)
// ---------------------------------------------------------------------------
__device__ __forceinline__ float gather_dt(const float* __restrict__ dt, int ix, int iy) {
    const bool v = (ix >= 0) & (ix < W_) & (iy >= 0) & (iy < H_);
    const int cx = min(max(ix, 0), W_ - 1);
    const int cy = min(max(iy, 0), H_ - 1);
    const float val = dt[cy * W_ + cx];
    return v ? val : 0.0f;
}

// ---------------------------------------------------------------------------
// Kernel 1: one streaming pass over all pixels of one image per blockIdx.y.
// Accumulates pull / penalty / benefit / fg-count (scalars) and per-instance
// embedding sums (LDS histogram -> global atomics).
// ws layout (floats): [0..3] = n_fg, pull, pen, ben ; [4 + (b*16+k)*3 + {0,1,2}]
//                     = sum_x, sum_y, count for instance k (label k+1) of batch b
// ---------------------------------------------------------------------------
__global__ __launch_bounds__(256) void skel_loss_main(
    const float* __restrict__ offsets,
    const int*   __restrict__ labels,
    const float* __restrict__ skel,
    const float* __restrict__ dtnorm,
    const float* __restrict__ dtgrad,
    float* __restrict__ ws)
{
    __shared__ float lseg[K_ * 3];
    __shared__ float lscal[4];
    const int tid = threadIdx.x;
    for (int i = tid; i < K_ * 3; i += blockDim.x) lseg[i] = 0.0f;
    if (tid < 4) lscal[tid] = 0.0f;
    __syncthreads();

    const int b = blockIdx.y;
    const float* __restrict__ offx = offsets + (size_t)b * (S_ * HW_);
    const float* __restrict__ offy = offx + HW_;
    const float* __restrict__ skx  = skel   + (size_t)b * (S_ * HW_);
    const float* __restrict__ sky  = skx + HW_;
    const float* __restrict__ gxp  = dtgrad + (size_t)b * (S_ * HW_);
    const float* __restrict__ gyp  = gxp + HW_;
    const float* __restrict__ dt   = dtnorm + (size_t)b * HW_;
    const int*   __restrict__ lab  = labels + (size_t)b * HW_;

    float aPull = 0.0f, aPen = 0.0f, aBen = 0.0f, aFg = 0.0f;

    const int nChunks = HW_ / 4;                     // W_ % 4 == 0: chunk never crosses a row
    const int stride  = gridDim.x * blockDim.x;
    for (int c = blockIdx.x * blockDim.x + tid; c < nChunks; c += stride) {
        const float4 ox4 = reinterpret_cast<const float4*>(offx)[c];
        const float4 oy4 = reinterpret_cast<const float4*>(offy)[c];
        const float4 sx4 = reinterpret_cast<const float4*>(skx)[c];
        const float4 sy4 = reinterpret_cast<const float4*>(sky)[c];
        const float4 gx4 = reinterpret_cast<const float4*>(gxp)[c];
        const float4 gy4 = reinterpret_cast<const float4*>(gyp)[c];
        const int4   lb4 = reinterpret_cast<const int4*>(lab)[c];

        const int p0 = c * 4;
        const int hrow = p0 / W_;
        const float hf = (float)hrow;
        const float wbase = (float)(p0 - hrow * W_);

        const float oxs[4] = {ox4.x, ox4.y, ox4.z, ox4.w};
        const float oys[4] = {oy4.x, oy4.y, oy4.z, oy4.w};
        const float sxs[4] = {sx4.x, sx4.y, sx4.z, sx4.w};
        const float sys[4] = {sy4.x, sy4.y, sy4.z, sy4.w};
        const float gxs[4] = {gx4.x, gx4.y, gx4.z, gx4.w};
        const float gys[4] = {gy4.x, gy4.y, gy4.z, gy4.w};
        const int   lbs[4] = {lb4.x, lb4.y, lb4.z, lb4.w};

        #pragma unroll
        for (int k = 0; k < 4; ++k) {
            const int lb = lbs[k];
            if (lb <= 0) continue;                   // all terms are masked by fg
            const float ox = oxs[k], oy = oys[k];
            const float ex = wbase + (float)k + ox;  // coords[0] = x = column index
            const float ey = hf + oy;                // coords[1] = y = row index

            // L_pull
            const float dx = ex - sxs[k];
            const float dy = ey - sys[k];
            aPull += dx * dx + dy * dy;

            // L_penalty: 1 - cos(offset, grad), l2_normalize with max(||.||, eps)
            const float gx = gxs[k], gy = gys[k];
            const float no = sqrtf(ox * ox + oy * oy);
            const float ng = sqrtf(gx * gx + gy * gy);
            aPen += 1.0f - (ox * gx + oy * gy) / (fmaxf(no, 1e-5f) * fmaxf(ng, 1e-5f));

            // L_benefit: bilinear sample of dt_norm at embedding (zeros padding)
            const float xf = floorf(ex), yf = floorf(ey);
            const float wx = ex - xf, wy = ey - yf;
            const int x0 = (int)xf, y0 = (int)yf;
            const float s = gather_dt(dt, x0,     y0    ) * (1.0f - wx) * (1.0f - wy)
                          + gather_dt(dt, x0 + 1, y0    ) * wx          * (1.0f - wy)
                          + gather_dt(dt, x0,     y0 + 1) * (1.0f - wx) * wy
                          + gather_dt(dt, x0 + 1, y0 + 1) * wx          * wy;
            aBen += 1.0f - s;
            aFg  += 1.0f;

            // per-instance embedding sums (labels 1..16 -> slot lb-1)
            const int li = (lb - 1) * 3;
            atomicAdd(&lseg[li + 0], ex);
            atomicAdd(&lseg[li + 1], ey);
            atomicAdd(&lseg[li + 2], 1.0f);
        }
    }

    // wave32 shuffle reduction of scalar partials
    #pragma unroll
    for (int off = 16; off > 0; off >>= 1) {
        aFg   += __shfl_down(aFg,   off, 32);
        aPull += __shfl_down(aPull, off, 32);
        aPen  += __shfl_down(aPen,  off, 32);
        aBen  += __shfl_down(aBen,  off, 32);
    }
    if ((tid & 31) == 0) {
        atomicAdd(&lscal[0], aFg);
        atomicAdd(&lscal[1], aPull);
        atomicAdd(&lscal[2], aPen);
        atomicAdd(&lscal[3], aBen);
    }
    __syncthreads();

    if (tid < 4)      atomicAdd(&ws[tid], lscal[tid]);
    if (tid < K_ * 3) atomicAdd(&ws[4 + b * (K_ * 3) + tid], lseg[tid]);
}

// ---------------------------------------------------------------------------
// Kernel 2: finalize. One wave per batch. The K x K pairwise dot-product
// (Gram) matrix of instance means M (16 x 2, zero-padded to K=4) is computed
// with V_WMMA_F32_16X16X4_F32: G = M * M^T in a single WMMA per batch.
// dist^2 = |mi|^2 + |mj|^2 - 2 Gij ; hinge + valid-pair masking follows.
// ---------------------------------------------------------------------------
__global__ __launch_bounds__(512) void skel_loss_final(
    const float* __restrict__ ws, float* __restrict__ out)
{
    __shared__ float smx[B_][K_];
    __shared__ float smy[B_][K_];
    __shared__ float snr[B_][K_];
    __shared__ float svl[B_][K_];
    __shared__ float spush[B_];

    const int tid  = threadIdx.x;
    const int b    = tid >> 5;   // wave index == batch index (16 waves)
    const int lane = tid & 31;

    if (lane < K_) {
        const float* sp = ws + 4 + (b * K_ + lane) * 3;
        const float cnt = sp[2];
        const float inv = 1.0f / fmaxf(cnt, 1.0f);
        const float x = sp[0] * inv;
        const float y = sp[1] * inv;
        smx[b][lane] = x;
        smy[b][lane] = y;
        snr[b][lane] = x * x + y * y;
        svl[b][lane] = (cnt > 0.0f) ? 1.0f : 0.0f;
    }
    __syncthreads();

    // A (16x4 f32): lanes 0-15 hold row M=lane, K=0 (vgpr0) / K=1 (vgpr1);
    // lanes 16-31 hold K=2/K=3 -> zero padding. B = A gives G = M * M^T.
    v8f c0 = {};
    v8f g;
#if __has_builtin(__builtin_amdgcn_wmma_f32_16x16x4_f32)
    v2f a;
    a.x = (lane < 16) ? smx[b][lane] : 0.0f;
    a.y = (lane < 16) ? smy[b][lane] : 0.0f;
    g = __builtin_amdgcn_wmma_f32_16x16x4_f32(false, a, false, a,
                                              (short)0, c0, false, false);
#else
    v16h ah;
    #pragma unroll
    for (int e = 0; e < 16; ++e) ah[e] = (_Float16)0.0f;
    if (lane < 16) {
        ah[0] = (_Float16)smx[b][lane];
        ah[1] = (_Float16)smy[b][lane];
    }
    g = __builtin_amdgcn_wmma_f32_16x16x32_f16(false, ah, false, ah,
                                               (short)0, c0, false, false);
#endif

    // D layout: VGPR r -> lanes 0-15: G[r][lane]; lanes 16-31: G[r+8][lane-16]
    float hsum = 0.0f;
    const int j     = lane & 15;
    const int ibase = (lane < 16) ? 0 : 8;
    #pragma unroll
    for (int r = 0; r < 8; ++r) {
        const int i = ibase + r;
        const float d2 = snr[b][i] + snr[b][j] - 2.0f * g[r];
        const float d  = sqrtf(fmaxf(d2, 0.0f) + 1e-12f);
        const float h  = fmaxf(20.0f - d, 0.0f);
        const float m  = svl[b][i] * svl[b][j] * ((i != j) ? 1.0f : 0.0f);
        hsum += h * h * m;
    }
    #pragma unroll
    for (int off = 16; off > 0; off >>= 1)
        hsum += __shfl_down(hsum, off, 32);

    if (lane == 0) {
        float nv = 0.0f;
        for (int k = 0; k < K_; ++k) nv += svl[b][k];
        const float npairs = fmaxf(nv * (nv - 1.0f), 1.0f);
        spush[b] = hsum / npairs;
    }
    __syncthreads();

    if (tid == 0) {
        float lpush = 0.0f;
        for (int i = 0; i < B_; ++i) lpush += spush[i];
        lpush *= (1.0f / (float)B_);
        const float nfg   = fmaxf(ws[0], 1.0f);
        const float lpull = ws[1] / nfg;
        const float lpen  = ws[2] / nfg;
        const float lben  = ws[3] / nfg;
        out[0] = 1.0f * lpull + 1.0f * lpush + 1.0f * lpen + 5.0f * lben;
        out[1] = lpull;
        out[2] = lpush;
        out[3] = lpen;
        out[4] = lben;
    }
}

// ---------------------------------------------------------------------------
extern "C" void kernel_launch(void* const* d_in, const int* in_sizes, int n_in,
                              void* d_out, int out_size, void* d_ws, size_t ws_size,
                              hipStream_t stream) {
    (void)in_sizes; (void)n_in; (void)out_size; (void)ws_size;
    const float* offsets = (const float*)d_in[0];
    const int*   labels  = (const int*)  d_in[1];
    const float* skel    = (const float*)d_in[2];
    const float* dtnorm  = (const float*)d_in[3];
    const float* dtgrad  = (const float*)d_in[4];
    float* ws = (float*)d_ws;

    // zero the accumulators every call (graph-capture safe)
    hipMemsetAsync(ws, 0, (4 + B_ * K_ * 3) * sizeof(float), stream);

    dim3 grid(256, B_, 1);                 // 256 blocks per image, one image per y
    skel_loss_main<<<grid, 256, 0, stream>>>(offsets, labels, skel, dtnorm, dtgrad, ws);
    skel_loss_final<<<1, 512, 0, stream>>>(ws, (float*)d_out);
}
